// GATInnerLayer_12077448036818
// MI455X (gfx1250) — matrix operand order repaired and checked by reading, not compiled, not run on hardware
//
#include <hip/hip_runtime.h>
#include <hip/hip_bf16.h>

typedef __attribute__((ext_vector_type(16))) _Float16 v16h;
typedef __attribute__((ext_vector_type(8)))  float    v8f;

#define DIM 64
#define NODES_PER_BLOCK 16

// ---------------------------------------------------------------------------
// Prep: convert Wq / Wk to f16 once per grid (so the attention kernel can
// DMA them straight into LDS with async b128 loads, no VGPR conversion).
// ---------------------------------------------------------------------------
__global__ void gat_prep_w_kernel(const float* __restrict__ Wq,
                                  const float* __restrict__ Wk,
                                  _Float16* __restrict__ wq16g,
                                  _Float16* __restrict__ wk16g)
{
    int i = blockIdx.x * blockDim.x + threadIdx.x;
    if (i < DIM * DIM) {
        wq16g[i] = (_Float16)Wq[i];
        wk16g[i] = (_Float16)Wk[i];
    }
}

// ---------------------------------------------------------------------------
// Fused kernel: per 16-node tile
//   stage h (f32) + f16 weights into LDS via global_load_async_to_lds_b128
//   wave0: q = h @ Wq^T via v_wmma_f32_16x16x32_f16  (8 WMMAs)
//   wave1: k = h @ Wk^T via v_wmma_f32_16x16x32_f16  (8 WMMAs)
//   then 64 threads do column-softmax attention and write a[node,:] to ws.
// ---------------------------------------------------------------------------
__global__ __launch_bounds__(64) void gat_attn_kernel(
    const float* __restrict__ h,
    const _Float16* __restrict__ wq16g,
    const _Float16* __restrict__ wk16g,
    float* __restrict__ a_out,
    int N)
{
    __shared__ __align__(16) _Float16 hf16[NODES_PER_BLOCK * DIM];   // 2 KB
    __shared__ __align__(16) float    hf32[NODES_PER_BLOCK * DIM];   // 4 KB
    __shared__ __align__(16) _Float16 wq16[DIM * DIM];               // 8 KB
    __shared__ __align__(16) _Float16 wk16[DIM * DIM];               // 8 KB
    __shared__ __align__(16) float    q_lds[NODES_PER_BLOCK * DIM];  // 4 KB
    __shared__ __align__(16) float    k_lds[NODES_PER_BLOCK * DIM];  // 4 KB
    __shared__ __align__(16) float    Esh[DIM * (DIM + 1)];          // 16.25 KB
    __shared__ float    Zr[DIM];

    const int tid = threadIdx.x;           // 0..63
    const int node_base = blockIdx.x * NODES_PER_BLOCK;

    // ---- async DMA: h tile (f32) into LDS, 16B chunks ----
    for (int c = tid; c < NODES_PER_BLOCK * DIM / 4; c += 64) {   // 256 float4s
        int row = c >> 4;
        int node = node_base + row;
        if (node >= N) node = N - 1;                // clamp: values unused, keep addr valid
        const float* g = h + (size_t)node * DIM + (c & 15) * 4;
        uint32_t l = (uint32_t)(uintptr_t)(hf32 + row * DIM + (c & 15) * 4);
        asm volatile("global_load_async_to_lds_b128 %0, %1, off"
                     :: "v"(l), "v"(g) : "memory");
    }
    // ---- async DMA: f16 weights into LDS (8 KB each, L2 broadcast) ----
    for (int c = tid; c < DIM * DIM / 8; c += 64) {               // 512 b128 chunks
        const _Float16* gq = wq16g + c * 8;
        uint32_t lq = (uint32_t)(uintptr_t)(wq16 + c * 8);
        asm volatile("global_load_async_to_lds_b128 %0, %1, off"
                     :: "v"(lq), "v"(gq) : "memory");
        const _Float16* gk = wk16g + c * 8;
        uint32_t lk = (uint32_t)(uintptr_t)(wk16 + c * 8);
        asm volatile("global_load_async_to_lds_b128 %0, %1, off"
                     :: "v"(lk), "v"(gk) : "memory");
    }
    asm volatile("s_wait_asynccnt 0x0" ::: "memory");
    __syncthreads();

    // ---- convert h tile to f16 for the WMMA A operand (LDS -> LDS) ----
    for (int t = tid; t < NODES_PER_BLOCK * DIM; t += 64) {
        hf16[t] = (_Float16)hf32[t];
    }
    __syncthreads();

    // ---- WMMA phase: wave0 -> q, wave1 -> k ----
    {
        const int lane = tid & 31;
        const int wv   = tid >> 5;         // 0: q, 1: k
        const int m    = lane & 15;        // row within tile / B column
        const int hi   = lane >> 4;        // half-wave select

        const _Float16* __restrict__ Wf16 = wv ? wk16 : wq16;
        float* __restrict__ OUT = wv ? k_lds : q_lds;

        for (int ct = 0; ct < 4; ++ct) {   // 4 column tiles of 16 features
            v8f acc = {};
            for (int kc = 0; kc < 2; ++kc) {  // K = 64 = 2 x 32
                v16h A, B;
                #pragma unroll
                for (int v = 0; v < 8; ++v) {
                    // A (16x32 f16): VGPR v holds K pair; v<4 -> K=2v, v>=4 -> K=16+2(v-4); +8*hi; +32*kc
                    int ka = ((v < 4) ? (2 * v) : (16 + 2 * (v - 4))) + 8 * hi + 32 * kc;
                    A[2 * v]     = hf16[m * DIM + ka];
                    A[2 * v + 1] = hf16[m * DIM + ka + 1];
                    // B (32x16 f16): same K order as A: v<4 -> K=2v, v>=4 -> K=16+2(v-4); +8*hi; +32*kc
                    int kb = ((v < 4) ? (2 * v) : (16 + 2 * (v - 4))) + 8 * hi + 32 * kc;
                    int feat = 16 * ct + m;            // output feature index (B column)
                    // q[n,i] = sum_k h[n,k] * W[i,k]  ->  B[kb][feat] = W[feat][kb]
                    B[2 * v]     = Wf16[feat * DIM + kb];
                    B[2 * v + 1] = Wf16[feat * DIM + kb + 1];
                }
                acc = __builtin_amdgcn_wmma_f32_16x16x32_f16(
                        /*neg_a=*/false, A, /*neg_b=*/false, B,
                        /*c_mod=*/(short)0, acc,
                        /*reuse_a=*/false, /*reuse_b=*/false);
            }
            // D (16x16 f32): lane n = lane&15 ; VGPR r -> M = r + 8*hi
            #pragma unroll
            for (int r = 0; r < 8; ++r) {
                OUT[(r + 8 * hi) * DIM + 16 * ct + m] = acc[r];
            }
        }
    }
    __syncthreads();

    // ---- softmax-attention phase: a[n,i] = sum_j softmax_i(q_i*k_j/8) * h_j ----
    const float scale = 0.125f;  // 1/sqrt(64)
    for (int local = 0; local < NODES_PER_BLOCK; ++local) {
        const float* qrow = q_lds + local * DIM;
        const float* krow = k_lds + local * DIM;

        // redundant per-thread max/min of q (broadcast LDS reads, cheap)
        float qmax = -3.4e38f, qmin = 3.4e38f;
        #pragma unroll 8
        for (int i = 0; i < DIM; ++i) {
            float qv = qrow[i];
            qmax = fmaxf(qmax, qv);
            qmin = fminf(qmin, qv);
        }

        // pass 1: thread j computes column j of E and Z_j
        {
            const int j = tid;
            float cj = krow[j] * scale;
            float mj = (cj >= 0.0f) ? (qmax * cj) : (qmin * cj);
            float Z = 0.0f;
            #pragma unroll 8
            for (int i = 0; i < DIM; ++i) {
                float e = __expf(qrow[i] * cj - mj);
                Esh[j * (DIM + 1) + i] = e;
                Z += e;
            }
            Zr[j] = 1.0f / Z;   // Z >= 1 (max term contributes exp(0))
        }
        __syncthreads();

        // pass 2: thread i reduces over j
        {
            const int i = tid;
            float acc = 0.0f;
            #pragma unroll 8
            for (int j = 0; j < DIM; ++j) {
                acc += Esh[j * (DIM + 1) + i] * Zr[j] * hf32[local * DIM + j];
            }
            int node = node_base + local;
            if (node < N) a_out[(size_t)node * DIM + i] = acc;
        }
        __syncthreads();   // protect Esh/Zr before next node iteration
    }
}

// ---------------------------------------------------------------------------
// Edge scatter: 16 threads per edge, float4 gather from a (L2-resident),
// f32 atomics into per-dst sums; one count atomic per edge.
// ---------------------------------------------------------------------------
__global__ void gat_scatter_kernel(
    const float* __restrict__ a,
    const int* __restrict__ src,
    const int* __restrict__ dst,
    float* __restrict__ sums,
    float* __restrict__ counts,
    int E)
{
    int idx = blockIdx.x * blockDim.x + threadIdx.x;
    int e = idx >> 4;
    if (e >= E) return;
    int chunk = idx & 15;
    int s = src[e];
    int d = dst[e];
    const float4 v = reinterpret_cast<const float4*>(a + (size_t)s * DIM)[chunk];
    float* base = sums + (size_t)d * DIM + chunk * 4;
    atomicAdd(base + 0, v.x);
    atomicAdd(base + 1, v.y);
    atomicAdd(base + 2, v.z);
    atomicAdd(base + 3, v.w);
    if (chunk == 0) atomicAdd(&counts[d], 1.0f);
}

// ---------------------------------------------------------------------------
// Finalize: out = sums / max(counts, 1)
// ---------------------------------------------------------------------------
__global__ void gat_finalize_kernel(
    const float* __restrict__ sums,
    const float* __restrict__ counts,
    float* __restrict__ out,
    int total)
{
    int i = blockIdx.x * blockDim.x + threadIdx.x;
    if (i < total) {
        out[i] = sums[i] / fmaxf(counts[i >> 6], 1.0f);
    }
}

extern "C" void kernel_launch(void* const* d_in, const int* in_sizes, int n_in,
                              void* d_out, int out_size, void* d_ws, size_t ws_size,
                              hipStream_t stream) {
    const float* h   = (const float*)d_in[0];
    const float* Wq  = (const float*)d_in[1];
    const float* Wk  = (const float*)d_in[2];
    const int*   src = (const int*)d_in[3];
    const int*   dst = (const int*)d_in[4];

    const int N = in_sizes[0] / DIM;
    const int E = in_sizes[3];

    float* out    = (float*)d_out;
    float* a      = (float*)d_ws;                       // N*64 f32
    float* sums   = a + (size_t)N * DIM;                // N*64 f32
    float* counts = sums + (size_t)N * DIM;             // N f32
    _Float16* wq16g = (_Float16*)(counts + N);          // 4096 f16 (16B-aligned: N*4 % 16 == 0 for N=30000)
    _Float16* wk16g = wq16g + DIM * DIM;                // 4096 f16

    // zero accumulators (sums + counts are contiguous)
    hipMemsetAsync(sums, 0, sizeof(float) * ((size_t)N * DIM + (size_t)N), stream);

    // convert weights to f16 once
    gat_prep_w_kernel<<<(DIM * DIM + 255) / 256, 256, 0, stream>>>(Wq, Wk, wq16g, wk16g);

    // fused q/k WMMA + softmax attention
    int nblk1 = (N + NODES_PER_BLOCK - 1) / NODES_PER_BLOCK;
    gat_attn_kernel<<<nblk1, 64, 0, stream>>>(h, wq16g, wk16g, a, N);

    // edge scatter-sum + counts
    long long scatter_threads = (long long)E * 16;
    int nblk2 = (int)((scatter_threads + 255) / 256);
    gat_scatter_kernel<<<nblk2, 256, 0, stream>>>(a, src, dst, sums, counts, E);

    // mean
    int total = N * DIM;
    int nblk3 = (total + 255) / 256;
    gat_finalize_kernel<<<nblk3, 256, 0, stream>>>(sums, counts, out, total);
}
